// PAM_Module_90701119357050
// MI455X (gfx1250) — compile-verified, matrix operand order checked by reading, not run on hardware
//
#include <hip/hip_runtime.h>
#include <cstddef>

// ---------------------------------------------------------------------------
// PAM (position attention) for MI455X / gfx1250, wave32, bf16 WMMA pipeline.
// B=4, C=512, C8=64, N=4096.  q=Wq@x, k=Wk@x, v=Wv@x (bf16 in workspace),
// then SINGLE-PASS flash attention: online softmax with accumulator rescale
// (no NxN tensor, QK^T computed exactly once).
// All LDS tiles stored K-contiguous so WMMA fragments are 2x 16-byte loads.
// ---------------------------------------------------------------------------

typedef __attribute__((ext_vector_type(16))) __bf16          v16bf;
typedef __attribute__((ext_vector_type(8)))  __bf16          v8bf;
typedef __attribute__((ext_vector_type(8)))  float           v8f;
typedef __attribute__((ext_vector_type(8)))  unsigned short  us8;

static constexpr int Bb  = 4;
static constexpr int C   = 512;
static constexpr int C8  = 64;
static constexpr int N   = 4096;

static constexpr int LDT = 72;  // ushort LDS row stride: mult. of 8 (16B aligned
                                // fragment runs), 144B row step = 36-bank stride
static constexpr int LSS = 65;  // float stride for S tile
static constexpr int LES = 17;  // float stride for epilogue transpose tile

// ---- bf16 <-> f32 helper (workspace/LDS stored as ushort bit patterns) ----
__device__ __forceinline__ unsigned short f_to_us(float f) {
  unsigned int u = __builtin_bit_cast(unsigned int, f);
  u += 0x7FFFu + ((u >> 16) & 1u);            // round-to-nearest-even
  return (unsigned short)(u >> 16);
}

// Fragment load per ISA 7.12.2 wave32 layout, K contiguous in memory:
// lane lid covers (row = rowbase + (lid&15),
//                  k = koff + 8*(lid>=16) + {0..7} and +16+{0..7}).
__device__ __forceinline__ v16bf frag_ld(const unsigned short* __restrict__ base,
                                         int ld, int rowbase, int koff, int lid) {
  const int rr = lid & 15;
  const int hb = (lid >> 4) << 3;
  const unsigned short* p = base + (size_t)(rowbase + rr) * ld + koff + hb;
  us8 lo = *(const us8*)(p);
  us8 hi = *(const us8*)(p + 16);
  v8bf a = __builtin_bit_cast(v8bf, lo);
  v8bf b = __builtin_bit_cast(v8bf, hi);
  return __builtin_shufflevector(a, b, 0, 1, 2, 3, 4, 5, 6, 7,
                                 8, 9, 10, 11, 12, 13, 14, 15);
}

#define WMMA_BF16(A, Bm, Cacc) \
  __builtin_amdgcn_wmma_f32_16x16x32_bf16(false, (A), false, (Bm), (short)0, (Cacc), false, false)

// ---------------------------------------------------------------------------
// Kernel 1: fused QKV projection.  grid = (N/64, 10, B), block = 256 (8 waves).
// blockIdx.y: 0 -> q rows 0..63, 1 -> k rows 0..63, 2..9 -> v rows (y-2)*64.
// ---------------------------------------------------------------------------
__global__ __launch_bounds__(256) void pam_proj_kernel(
    const float* __restrict__ x,
    const float* __restrict__ Wq, const float* __restrict__ bq,
    const float* __restrict__ Wk, const float* __restrict__ bk,
    const float* __restrict__ Wv, const float* __restrict__ bv,
    unsigned short* __restrict__ qws, unsigned short* __restrict__ kws,
    unsigned short* __restrict__ vws) {
  const int tid  = threadIdx.x;
  const int wave = tid >> 5;
  const int lid  = tid & 31;
  const int b    = blockIdx.z;
  const int n0   = blockIdx.x * 64;
  const int y    = blockIdx.y;

  const float* Wsel;
  const float* bsel;
  unsigned short* dsel;
  int obase;
  if (y == 0)      { Wsel = Wq; bsel = bq; dsel = qws + (size_t)b * C8 * N; obase = 0; }
  else if (y == 1) { Wsel = Wk; bsel = bk; dsel = kws + (size_t)b * C8 * N; obase = 0; }
  else             { Wsel = Wv; bsel = bv; dsel = vws + (size_t)b * C  * N; obase = (y - 2) * 64; }

  __shared__ __align__(16) unsigned short Wt[64 * LDT];   // [o][c]  (K=c contiguous)
  __shared__ __align__(16) unsigned short xt[64 * LDT];   // [n][c]  (K=c contiguous)

  const int cc = tid & 63;   // coalesced global column (invariant)
  const int r0 = tid >> 6;   // starting row (invariant, step 4)

  const int strip = wave >> 1;
  const int tcb   = (wave & 1) << 1;
  v8f acc0 = {};
  v8f acc1 = {};

  for (int c0 = 0; c0 < C; c0 += 64) {
    __syncthreads();
    {
      const float* wg = Wsel + (size_t)(obase + r0) * C + c0 + cc;
      const float* xg = x + (((size_t)b * C) + c0 + r0) * N + n0 + cc;
      unsigned short* wl = &Wt[r0 * LDT + cc];
      unsigned short* xl = &xt[cc * LDT + r0];       // transposed store
#pragma unroll
      for (int it = 0; it < 16; ++it) {
        wl[it * 4 * LDT] = f_to_us(wg[(size_t)it * 4 * C]);
        xl[it * 4]       = f_to_us(xg[(size_t)it * 4 * N]);
      }
    }
    __syncthreads();
#pragma unroll
    for (int kk = 0; kk < 2; ++kk) {
      v16bf a  = frag_ld(Wt, LDT, strip * 16, kk * 32, lid);        // A[o][c]
      v16bf b0 = frag_ld(xt, LDT, tcb * 16, kk * 32, lid);          // B[c][n]=xt[n][c]
      v16bf b1 = frag_ld(xt, LDT, (tcb + 1) * 16, kk * 32, lid);
      acc0 = WMMA_BF16(a, b0, acc0);
      acc1 = WMMA_BF16(a, b1, acc1);
    }
  }

  const int rbase = strip * 16 + ((lid >> 4) << 3);
  const int col   = lid & 15;
#pragma unroll
  for (int r = 0; r < 8; ++r) {
    const int orow = obase + rbase + r;
    const float bias = bsel[orow];
    dsel[(size_t)orow * N + n0 + tcb * 16 + col]       = f_to_us(acc0[r] + bias);
    dsel[(size_t)orow * N + n0 + (tcb + 1) * 16 + col] = f_to_us(acc1[r] + bias);
  }
}

// ---------------------------------------------------------------------------
// Kernel 2: single-pass flash attention + output.  grid = (N/64, B), block 256.
// Wave w: n-strip (w>>1), S col-tiles 2*(w&1)..+1, O c-range (w&1)*256..+255.
// Online softmax: per m-tile update running max, rescale accumulators by
// exp(old_max - new_max), accumulate unnormalized O and row sums.
// ---------------------------------------------------------------------------
__global__ __launch_bounds__(256) void pam_attn_kernel(
    const unsigned short* __restrict__ qws, const unsigned short* __restrict__ kws,
    const unsigned short* __restrict__ vws, const float* __restrict__ x,
    const float* __restrict__ gamma, float* __restrict__ out) {
  const int tid  = threadIdx.x;
  const int wave = tid >> 5;
  const int lid  = tid & 31;
  const int b    = blockIdx.y;
  const int n0   = blockIdx.x * 64;

  const unsigned short* qb = qws + (size_t)b * C8 * N;
  const unsigned short* kb = kws + (size_t)b * C8 * N;
  const unsigned short* vb = vws + (size_t)b * C * N;

  __shared__ __align__(16) unsigned short qs[64 * LDT];   // [n][o] (K=o contiguous)
  __shared__ __align__(16) unsigned short ks[64 * LDT];   // [m][o] (K=o contiguous)
  __shared__ __align__(16) unsigned short ps[64 * LDT];   // [n][m] (K=m contiguous)
  __shared__ float Ss[64 * LSS];                          // [n][m] f32
  __shared__ float red[64 * 4];                           // per-row partials
  __shared__ float es[8 * 16 * LES];                      // per-wave epilogue tile
  __shared__ float rmax[64];                              // running row max
  __shared__ float rsum[64];                              // running row sum
  __shared__ float rfac[64];                              // this-tile rescale

  const int cc = tid & 63;   // coalesced column (invariant)
  const int r0 = tid >> 6;   // starting row (invariant, step 4)

  const int strip = wave >> 1;              // n strip (16 rows)
  const int tcb   = (wave & 1) << 1;        // S col tiles
  const int cbase = (wave & 1) * 256;       // output channel range

  // softmax thread mapping: 4 threads per row, 16 cols each
  const int srow = tid >> 2;
  const int sq   = tid & 3;

  // load q tile once: q stored [o][n] in ws -> qs[n][o]
  {
    const unsigned short* gq = qb + (size_t)r0 * N + n0 + cc;
    unsigned short* lq = &qs[cc * LDT + r0];
#pragma unroll
    for (int it = 0; it < 16; ++it) lq[it * 4] = gq[(size_t)it * 4 * N];
  }
  if (tid < 64) { rmax[tid] = -3.0e38f; rsum[tid] = 0.0f; }
  __syncthreads();

  // Q A-fragments are invariant over the whole kernel: hoist.
  const v16bf aq0 = frag_ld(qs, LDT, strip * 16, 0, lid);
  const v16bf aq1 = frag_ld(qs, LDT, strip * 16, 32, lid);

  v8f oacc[16];
#pragma unroll
  for (int t = 0; t < 16; ++t) { v8f z = {}; oacc[t] = z; }

  const int rowt = (lid >> 4) << 3;   // fragment row base within tile (0/8)

  // ---------------- single pass over key tiles ----------------
  for (int m0 = 0; m0 < N; m0 += 64) {
    __syncthreads();            // prev PV readers of ps / rfac done
    {
      const unsigned short* gk = kb + (size_t)r0 * N + m0 + cc;
      unsigned short* lk = &ks[cc * LDT + r0];
#pragma unroll
      for (int it = 0; it < 16; ++it) lk[it * 4] = gk[(size_t)it * 4 * N];
    }
    if (tid < 64 && (m0 + 64) < N)
      __builtin_prefetch(&kb[(size_t)tid * N + m0 + 64], 0, 1);
    __syncthreads();

    // S = qT k for this 64x64 tile
#pragma unroll
    for (int t = 0; t < 2; ++t) {
      const int tc = tcb + t;
      v8f s = {};
      s = WMMA_BF16(aq0, frag_ld(ks, LDT, tc * 16, 0, lid), s);
      s = WMMA_BF16(aq1, frag_ld(ks, LDT, tc * 16, 32, lid), s);
      const int rb = strip * 16 + rowt;
      const int col = tc * 16 + (lid & 15);
#pragma unroll
      for (int r = 0; r < 8; ++r) Ss[(rb + r) * LSS + col] = s[r];
    }
    __syncthreads();

    // per-row tile max partials
    {
      const float* sr = &Ss[srow * LSS + sq * 16];
      float mx = -3.0e38f;
#pragma unroll
      for (int j = 0; j < 16; ++j) mx = fmaxf(mx, sr[j]);
      red[tid] = mx;
    }
    __syncthreads();

    // update running max, compute rescale factor, pre-scale running sum
    if (tid < 64) {
      const float* pr = &red[tid * 4];
      const float tmax = fmaxf(fmaxf(pr[0], pr[1]), fmaxf(pr[2], pr[3]));
      const float omax = rmax[tid];
      const float nmax = fmaxf(omax, tmax);
      const float fac  = __expf(omax - nmax);   // 0 on first tile
      rmax[tid] = nmax;
      rfac[tid] = fac;
      rsum[tid] *= fac;
    }
    __syncthreads();

    // P = exp(S - new_max) -> bf16, partial row sums
    {
      const float mx = rmax[srow];
      float acc = 0.0f;
#pragma unroll
      for (int j = 0; j < 16; ++j) {
        const int colx = sq * 16 + j;
        const float e = __expf(Ss[srow * LSS + colx] - mx);
        ps[srow * LDT + colx] = f_to_us(e);
        acc += e;
      }
      red[tid] = acc;
    }
    __syncthreads();
    if (tid < 64) {
      const float* pr = &red[tid * 4];
      rsum[tid] += (pr[0] + pr[1]) + (pr[2] + pr[3]);
    }

    // rescale accumulators by per-row factor (cheap VALU, co-executes w/ WMMA)
    float fr[8];
#pragma unroll
    for (int r = 0; r < 8; ++r) fr[r] = rfac[strip * 16 + rowt + r];
#pragma unroll
    for (int t = 0; t < 16; ++t)
#pragma unroll
      for (int r = 0; r < 8; ++r) oacc[t][r] *= fr[r];

    // O[n][c] += P[n][m] * v[c][m]; v fragments straight from global
    const unsigned short* vt = vb + m0;
    v16bf a0 = frag_ld(ps, LDT, strip * 16, 0, lid);    // A[n][m]
    v16bf a1 = frag_ld(ps, LDT, strip * 16, 32, lid);
#pragma unroll
    for (int t = 0; t < 16; ++t) {
      const int c0 = cbase + t * 16;
      oacc[t] = WMMA_BF16(a0, frag_ld(vt, N, c0, 0, lid), oacc[t]);
      oacc[t] = WMMA_BF16(a1, frag_ld(vt, N, c0, 32, lid), oacc[t]);
    }
  }

  __syncthreads();
  if (tid < 64) rsum[tid] = 1.0f / rsum[tid];   // one rcp per row
  __syncthreads();

  // ------- epilogue: transpose tiles through LDS -> coalesced 32B runs -------
  const float g      = gamma[0];
  const int   colt   = lid & 15;          // fragment col (c) in tile
  const int   c_row  = lid >> 1;          // writing phase: c row 0..15
  const int   nh     = (lid & 1) * 8;     // writing phase: n offset 0/8
  float* ebuf = &es[wave * 16 * LES];
#pragma unroll
  for (int t = 0; t < 16; ++t) {
#pragma unroll
    for (int r = 0; r < 8; ++r)
      ebuf[colt * LES + rowt + r] = oacc[t][r];   // es[c][n] transpose
    __syncthreads();
    const int cg = cbase + t * 16 + c_row;
    const size_t basei = ((size_t)b * C + cg) * N + n0 + strip * 16 + nh;
#pragma unroll
    for (int j = 0; j < 8; ++j) {
      const float val = ebuf[c_row * LES + nh + j] * rsum[strip * 16 + nh + j];
      out[basei + j] = g * val + x[basei + j];
    }
    __syncthreads();
  }
}

// ---------------------------------------------------------------------------
extern "C" void kernel_launch(void* const* d_in, const int* in_sizes, int n_in,
                              void* d_out, int out_size, void* d_ws, size_t ws_size,
                              hipStream_t stream) {
  (void)in_sizes; (void)n_in; (void)out_size; (void)ws_size;
  const float* x     = (const float*)d_in[0];
  const float* Wq    = (const float*)d_in[1];
  const float* bq    = (const float*)d_in[2];
  const float* Wk    = (const float*)d_in[3];
  const float* bk    = (const float*)d_in[4];
  const float* Wv    = (const float*)d_in[5];
  const float* bv    = (const float*)d_in[6];
  const float* gamma = (const float*)d_in[7];
  float* out = (float*)d_out;

  unsigned short* qws = (unsigned short*)d_ws;
  unsigned short* kws = qws + (size_t)Bb * C8 * N;
  unsigned short* vws = kws + (size_t)Bb * C8 * N;

  pam_proj_kernel<<<dim3(N / 64, 10, Bb), 256, 0, stream>>>(
      x, Wq, bq, Wk, bk, Wv, bv, qws, kws, vws);
  pam_attn_kernel<<<dim3(N / 64, Bb), 256, 0, stream>>>(
      qws, kws, vws, x, gamma, out);
}
